// MultiHeadAttention_14611478741655
// MI455X (gfx1250) — matrix operand order repaired and checked by reading, not compiled, not run on hardware
//
#include <hip/hip_runtime.h>
#include <hip/hip_bf16.h>

#define EMB   1024
#define NQKV  3072
#define SEQ   2048
#define BATCH 2
#define HEADS 16
#define HD    64
#define TOK   (BATCH * SEQ)        // 4096
#define BH    (BATCH * HEADS)      // 32
#define SCALE 0.125f               // (EMB/HEADS)^-0.5 = 64^-0.5

typedef __attribute__((ext_vector_type(16))) __bf16 bf16x16;
typedef __attribute__((ext_vector_type(8)))  __bf16 bf16x8;
typedef __attribute__((ext_vector_type(8)))  float  f32x8;
typedef __attribute__((ext_vector_type(4)))  unsigned int u32x4;
typedef __attribute__((ext_vector_type(8)))  int  i32x8;
typedef __attribute__((ext_vector_type(4)))  int  i32x4;

// ---- WMMA fragment loaders (wave32, v_wmma_f32_16x16x32_bf16 layouts) ----
// A 16x32 (MxK): lane L -> row M = L%16 ; elems 0..7 -> K=(L/16)*8+e ;
//                elems 8..15 -> K=16+(L/16)*8+(e-8)
__device__ __forceinline__ bf16x16 load_a_frag(const __bf16* base, int ld, int lane) {
    const __bf16* r = base + (size_t)(lane & 15) * ld + ((lane >> 4) << 3);
    bf16x8 lo = *(const bf16x8*)(r);
    bf16x8 hi = *(const bf16x8*)(r + 16);
    bf16x16 a;
#pragma unroll
    for (int i = 0; i < 8; ++i) { a[i] = lo[i]; a[i + 8] = hi[i]; }
    return a;
}
// B 32x16 (KxN), stored column-major-in-K (i.e. colbase[n*ld + k]):
// lane L -> col N = L%16 ; elems 0..7 -> K=(L/16)*8+e ;
//                elems 8..15 -> K=16+(L/16)*8+(e-8)
__device__ __forceinline__ bf16x16 load_b_frag(const __bf16* colbase, int ld, int lane) {
    const __bf16* r = colbase + (size_t)(lane & 15) * ld + ((lane >> 4) << 3);
    bf16x8 lo = *(const bf16x8*)(r);
    bf16x8 hi = *(const bf16x8*)(r + 16);
    bf16x16 b;
#pragma unroll
    for (int i = 0; i < 8; ++i) { b[i] = lo[i]; b[i + 8] = hi[i]; }
    return b;
}

__device__ __forceinline__ f32x8 wmma_bf16(bf16x16 a, bf16x16 b, f32x8 c) {
    return __builtin_amdgcn_wmma_f32_16x16x32_bf16(false, a, false, b, (short)0, c, false, false);
}

// ---- Tensor Data Mover: 2-D tile Global->LDS (bf16 elements) ----
// D# per CDNA5 ISA ch.8: group0 {count=1, lds_addr, global_addr(57b), type=2},
// group1 {data_size=1(2B), tensor_dim0/1, tile_dim0/1, tensor_dim0_stride}.
// dims/strides in elements. Groups 2/3 zero (2-D tensor).
// This toolchain exposes the 6-arg builtin: (g0, g1, g2, g3, g4, cpol).
__device__ __forceinline__ void tdm_load_2d_bf16(
    unsigned lds_off, const __bf16* gaddr,
    unsigned dim0, unsigned dim1, unsigned stride0,
    unsigned tile0, unsigned tile1)
{
    unsigned long long ga = (unsigned long long)(uintptr_t)gaddr;
    u32x4 g0;
    g0[0] = 1u;                                              // count=1
    g0[1] = lds_off;                                         // lds_addr (bytes)
    g0[2] = (unsigned)ga;                                    // global_addr[31:0]
    g0[3] = (unsigned)((ga >> 32) & 0x1FFFFFFull) | (2u << 30); // addr[56:32] | type=2
    i32x8 g1;
    g1[0] = (int)(1u << 16);                                 // wg_mask=0, data_size=1 (2B)
    g1[1] = (int)((dim0 & 0xFFFFu) << 16);                   // atomic_addr=0 | dim0 lo16
    g1[2] = (int)((dim0 >> 16) | ((dim1 & 0xFFFFu) << 16));  // dim0 hi | dim1 lo
    g1[3] = (int)((dim1 >> 16) | (tile0 << 16));             // dim1 hi | tile_dim0
    g1[4] = (int)(tile1 & 0xFFFFu);                          // tile_dim1 | tile_dim2=0
    g1[5] = (int)stride0;                                    // stride0[31:0]
    g1[6] = 0;                                               // stride0 hi16 | stride1 lo16
    g1[7] = 0;                                               // stride1 hi
    i32x4 z4 = {0, 0, 0, 0};
    i32x8 z8 = {0, 0, 0, 0, 0, 0, 0, 0};
    __builtin_amdgcn_tensor_load_to_lds(g0, g1, z4, z4, z8, 0);
}

// =====================================================================
// Kernel 1: qkv = x @ w_qkv + b_qkv, de-interleaved into Q,K,[V^T] bf16
// grid (NQKV/128, TOK/64), 256 threads (8 waves, 2x4, each 32x32 output)
// =====================================================================
__global__ __launch_bounds__(256) void qkv_gemm_kernel(
    const float* __restrict__ x, const float* __restrict__ w,
    const float* __restrict__ bias,
    __bf16* __restrict__ q_ws, __bf16* __restrict__ k_ws, __bf16* __restrict__ v_ws)
{
    __shared__ __align__(32) __bf16 lds_a[64][32];
    __shared__ __align__(32) __bf16 lds_b[128][32];   // [n][k] (transposed)

    const int tid = threadIdx.x, lane = tid & 31, wave = tid >> 5;
    const int wm = wave >> 2, wn = wave & 3;          // 2x4 waves
    const int mBlk = blockIdx.y * 64, nBlk = blockIdx.x * 128;

    f32x8 c[2][2] = {};
    for (int k0 = 0; k0 < EMB; k0 += 32) {
        __syncthreads();
        // A tile: 64x32 fp32 -> bf16
#pragma unroll
        for (int p = 0; p < 2; ++p) {
            int t = tid + p * 256;                    // 0..511 float4 slots
            int row = t >> 3, c4 = (t & 7) << 2;
            float4 v = *(const float4*)(x + (size_t)(mBlk + row) * EMB + k0 + c4);
            lds_a[row][c4 + 0] = (__bf16)v.x; lds_a[row][c4 + 1] = (__bf16)v.y;
            lds_a[row][c4 + 2] = (__bf16)v.z; lds_a[row][c4 + 3] = (__bf16)v.w;
        }
        // B tile: 32x128 fp32, stored transposed [n][k]
#pragma unroll
        for (int p = 0; p < 4; ++p) {
            int t = tid + p * 256;                    // 0..1023 float4 slots
            int kr = t >> 5, c4 = (t & 31) << 2;
            float4 v = *(const float4*)(w + (size_t)(k0 + kr) * NQKV + nBlk + c4);
            lds_b[c4 + 0][kr] = (__bf16)v.x; lds_b[c4 + 1][kr] = (__bf16)v.y;
            lds_b[c4 + 2][kr] = (__bf16)v.z; lds_b[c4 + 3][kr] = (__bf16)v.w;
        }
        __syncthreads();
#pragma unroll
        for (int tm = 0; tm < 2; ++tm) {
            bf16x16 a = load_a_frag(&lds_a[wm * 32 + tm * 16][0], 32, lane);
#pragma unroll
            for (int tn = 0; tn < 2; ++tn) {
                bf16x16 b = load_b_frag(&lds_b[wn * 32 + tn * 16][0], 32, lane);
                c[tm][tn] = wmma_bf16(a, b, c[tm][tn]);
            }
        }
    }
    // epilogue: bias + de-interleave (reshape ...,16,64,3), branchless
#pragma unroll
    for (int tm = 0; tm < 2; ++tm) {
        const int mBase = mBlk + wm * 32 + tm * 16 + ((lane >> 4) << 3);
        const int bb = mBase >> 11;               // constant within 64-row block
        const int tok0 = mBase & (SEQ - 1);
#pragma unroll
        for (int tn = 0; tn < 2; ++tn) {
            unsigned n = (unsigned)(nBlk + wn * 32 + tn * 16 + (lane & 15));
            unsigned h = n / 192u, rr = n % 192u, d = rr / 3u, which = rr % 3u;
            float bv = bias[n];
            int bh = bb * HEADS + (int)h;
            size_t idx_rm = ((size_t)bh * SEQ + tok0) * HD + d;  // q/k row-major
            size_t idx_tr = ((size_t)bh * HD + d) * SEQ + tok0;  // v transposed
            __bf16* bp  = (which == 0) ? q_ws : ((which == 1) ? k_ws : v_ws);
            size_t  idx = (which == 2) ? idx_tr : idx_rm;
            size_t  stp = (which == 2) ? (size_t)1 : (size_t)HD;
#pragma unroll
            for (int r = 0; r < 8; ++r)
                bp[idx + (size_t)r * stp] = (__bf16)(c[tm][tn][r] + bv);
        }
    }
}

// =====================================================================
// Kernel 2: streaming-softmax attention. grid (SEQ/128, BH), 256 threads;
// each wave owns 16 query rows; K/V staged to LDS by the Tensor Data
// Mover in 64-key chunks (wave 0 issues, TENSORcnt + barrier publishes).
// =====================================================================
__global__ __launch_bounds__(256) void attn_kernel(
    const __bf16* __restrict__ q_ws, const __bf16* __restrict__ k_ws,
    const __bf16* __restrict__ v_ws, __bf16* __restrict__ o_ws)
{
    __shared__ __align__(32) __bf16 k_tile[64][HD];   // [key][d]
    __shared__ __align__(32) __bf16 v_tile[HD][64];   // [d][key]
    __shared__ __align__(32) __bf16 p_lds[8][16][32];

    const int tid = threadIdx.x, lane = tid & 31, wave = tid >> 5;
    const int bh = blockIdx.y;
    const int b = bh >> 4, h = bh & 15;
    const int qRow = blockIdx.x * 128 + wave * 16;

    const __bf16* qg = q_ws + ((size_t)bh * SEQ + qRow) * HD;
    const __bf16* kg = k_ws + (size_t)bh * SEQ * HD;
    const __bf16* vg = v_ws + (size_t)bh * HD * SEQ;

    const unsigned k_off = (unsigned)(uintptr_t)&k_tile[0][0];
    const unsigned v_off = (unsigned)(uintptr_t)&v_tile[0][0];

    bf16x16 aq0 = load_a_frag(qg,      HD, lane);   // k-dims 0..31
    bf16x16 aq1 = load_a_frag(qg + 32, HD, lane);   // k-dims 32..63

    f32x8 o[4] = {};
    float mrow[8], lrow[8], scl[8];
#pragma unroll
    for (int r = 0; r < 8; ++r) { mrow[r] = -1e30f; lrow[r] = 0.f; }

    for (int kb = 0; kb < SEQ; kb += 64) {
        __syncthreads();                              // tiles free to overwrite
        if (wave == 0) {
            // K chunk: tensor [SEQ][HD] row-major; tile 64 keys x 64 d
            tdm_load_2d_bf16(k_off, kg + (size_t)kb * HD, HD, SEQ, HD, HD, 64);
            // V chunk: tensor [HD][SEQ] row-major; tile 64 keys (x) x 64 d (y)
            tdm_load_2d_bf16(v_off, vg + kb, SEQ, HD, SEQ, 64, HD);
            __builtin_amdgcn_s_wait_tensorcnt((short)0);
        }
        __syncthreads();                              // publish LDS tiles

#pragma unroll
        for (int kb2 = 0; kb2 < 64; kb2 += 32) {
            // S = Q K^T for 16 rows x 32 keys (two 16-col tiles), from LDS
            bf16x16 bk0a = load_b_frag(&k_tile[kb2][0],       HD, lane);
            bf16x16 bk0b = load_b_frag(&k_tile[kb2][32],      HD, lane);
            bf16x16 bk1a = load_b_frag(&k_tile[kb2 + 16][0],  HD, lane);
            bf16x16 bk1b = load_b_frag(&k_tile[kb2 + 16][32], HD, lane);
            f32x8 s0 = {}, s1 = {};
            s0 = wmma_bf16(aq0, bk0a, s0); s0 = wmma_bf16(aq1, bk0b, s0);
            s1 = wmma_bf16(aq0, bk1a, s1); s1 = wmma_bf16(aq1, bk1b, s1);

            // online softmax (rows across 16-lane halves; row = (lane/16)*8+r)
#pragma unroll
            for (int r = 0; r < 8; ++r) {
                float mx = fmaxf(s0[r], s1[r]);
#pragma unroll
                for (int msk = 8; msk >= 1; msk >>= 1) mx = fmaxf(mx, __shfl_xor(mx, msk, 32));
                float mnew = fmaxf(mrow[r], mx);
                float sc = __expf(mrow[r] - mnew);
                float e0 = __expf(s0[r] - mnew);
                float e1 = __expf(s1[r] - mnew);
                float ls = e0 + e1;
#pragma unroll
                for (int msk = 8; msk >= 1; msk >>= 1) ls += __shfl_xor(ls, msk, 32);
                lrow[r] = lrow[r] * sc + ls;
                mrow[r] = mnew;
                scl[r] = sc;
                int prow = ((lane >> 4) << 3) + r;
                p_lds[wave][prow][lane & 15]        = (__bf16)e0;
                p_lds[wave][prow][16 + (lane & 15)] = (__bf16)e1;
            }
#pragma unroll
            for (int t = 0; t < 4; ++t)
#pragma unroll
                for (int r = 0; r < 8; ++r) o[t][r] *= scl[r];

            asm volatile("s_wait_dscnt 0" ::: "memory");
            bf16x16 ap = load_a_frag(&p_lds[wave][0][0], 32, lane);
#pragma unroll
            for (int t = 0; t < 4; ++t) {
                bf16x16 bv = load_b_frag(&v_tile[t * 16][kb2], 64, lane);
                o[t] = wmma_bf16(ap, bv, o[t]);
            }
        }
    }
    // epilogue: softmax-then-SCALE; write O as (b, n, h, d) row-major bf16
#pragma unroll
    for (int t = 0; t < 4; ++t)
#pragma unroll
        for (int r = 0; r < 8; ++r) {
            int row = qRow + ((lane >> 4) << 3) + r;
            int d = t * 16 + (lane & 15);
            float val = o[t][r] * (SCALE / lrow[r]);
            o_ws[((size_t)(b * SEQ + row)) * EMB + h * HD + d] = (__bf16)val;
        }
}

// =====================================================================
// Kernel 3: out = O @ w_proj + b_proj (fp32 output)
// grid (EMB/128, TOK/64), 256 threads
// =====================================================================
__global__ __launch_bounds__(256) void proj_kernel(
    const __bf16* __restrict__ o_ws, const float* __restrict__ w,
    const float* __restrict__ bias, float* __restrict__ out)
{
    __shared__ __align__(32) __bf16 lds_a[64][32];
    __shared__ __align__(32) __bf16 lds_b[128][32];

    const int tid = threadIdx.x, lane = tid & 31, wave = tid >> 5;
    const int wm = wave >> 2, wn = wave & 3;
    const int mBlk = blockIdx.y * 64, nBlk = blockIdx.x * 128;

    f32x8 c[2][2] = {};
    for (int k0 = 0; k0 < EMB; k0 += 32) {
        __syncthreads();
        {   // A tile (already bf16): 64x32, one bf16x8 per thread
            int row = tid >> 2, c8 = (tid & 3) << 3;
            bf16x8 v = *(const bf16x8*)(o_ws + (size_t)(mBlk + row) * EMB + k0 + c8);
            *(bf16x8*)&lds_a[row][c8] = v;
        }
#pragma unroll
        for (int p = 0; p < 4; ++p) {
            int t = tid + p * 256;
            int kr = t >> 5, c4 = (t & 31) << 2;
            float4 v = *(const float4*)(w + (size_t)(k0 + kr) * EMB + nBlk + c4);
            lds_b[c4 + 0][kr] = (__bf16)v.x; lds_b[c4 + 1][kr] = (__bf16)v.y;
            lds_b[c4 + 2][kr] = (__bf16)v.z; lds_b[c4 + 3][kr] = (__bf16)v.w;
        }
        __syncthreads();
#pragma unroll
        for (int tm = 0; tm < 2; ++tm) {
            bf16x16 a = load_a_frag(&lds_a[wm * 32 + tm * 16][0], 32, lane);
#pragma unroll
            for (int tn = 0; tn < 2; ++tn) {
                bf16x16 b = load_b_frag(&lds_b[wn * 32 + tn * 16][0], 32, lane);
                c[tm][tn] = wmma_bf16(a, b, c[tm][tn]);
            }
        }
    }
#pragma unroll
    for (int tm = 0; tm < 2; ++tm)
#pragma unroll
        for (int tn = 0; tn < 2; ++tn) {
            int n = nBlk + wn * 32 + tn * 16 + (lane & 15);
            float bv = bias[n];
#pragma unroll
            for (int r = 0; r < 8; ++r) {
                int m = mBlk + wm * 32 + tm * 16 + ((lane >> 4) << 3) + r;
                out[(size_t)m * EMB + n] = c[tm][tn][r] + bv;
            }
        }
}

extern "C" void kernel_launch(void* const* d_in, const int* in_sizes, int n_in,
                              void* d_out, int out_size, void* d_ws, size_t ws_size,
                              hipStream_t stream) {
    const float* x      = (const float*)d_in[0];
    const float* w_qkv  = (const float*)d_in[1];
    const float* b_qkv  = (const float*)d_in[2];
    const float* w_proj = (const float*)d_in[3];
    const float* b_proj = (const float*)d_in[4];
    float* out = (float*)d_out;

    const size_t qkv_elems = (size_t)BH * SEQ * HD;   // 4 Mi elems each
    __bf16* q_ws = (__bf16*)d_ws;
    __bf16* k_ws = q_ws + qkv_elems;
    __bf16* v_ws = k_ws + qkv_elems;                  // stored transposed [bh][d][n]
    __bf16* o_ws = v_ws + qkv_elems;                  // [tok][emb]

    dim3 blk(256);
    dim3 g1(NQKV / 128, TOK / 64);
    qkv_gemm_kernel<<<g1, blk, 0, stream>>>(x, w_qkv, b_qkv, q_ws, k_ws, v_ws);

    dim3 g2(SEQ / 128, BH);
    attn_kernel<<<g2, blk, 0, stream>>>(q_ws, k_ws, v_ws, o_ws);

    dim3 g3(EMB / 128, TOK / 64);
    proj_kernel<<<g3, blk, 0, stream>>>(o_ws, w_proj, b_proj, out);
}
